// AdvancedMeshEncoder_80247168959174
// MI455X (gfx1250) — matrix operand order repaired and checked by reading, not compiled, not run on hardware
//
#include <hip/hip_runtime.h>
#include <hip/hip_bf16.h>

// ---------------------------------------------------------------------------
// AdvancedMeshEncoder forward for MI455X (gfx1250, wave32, WMMA).
// Heavy GEMMs (EdgeConv 2x[800K x 128 x 128], node GEMMs) run on
// v_wmma_f32_16x16x32_bf16 with fragment-ready LDS layouts so every
// fragment load is ds_load_b128 (no scalar u16 LDS traffic).
// Irregular graph ops (softmax segments, segment max, pooling) use L2 atomics.
// ---------------------------------------------------------------------------

typedef __bf16 bf16_t;
typedef __attribute__((ext_vector_type(16))) __bf16 v16bf;  // 32B, align 32
typedef __attribute__((ext_vector_type(8)))  __bf16 v8bf;   // 16B, align 16
typedef __attribute__((ext_vector_type(8)))  float  v8f;

#define N_NODES 50000
#define N_EDGES 800000
#define N_GRAPH 64

static __device__ __forceinline__ float lrelu(float x) { return x > 0.f ? x : 0.2f * x; }

// Order-preserving float<->uint encoding so atomicMax(uint) == float max.
static __device__ __forceinline__ unsigned ord_enc(float f) {
  unsigned u = __float_as_uint(f);
  return (u & 0x80000000u) ? ~u : (u | 0x80000000u);
}
static __device__ __forceinline__ float ord_dec(unsigned u) {
  return (u & 0x80000000u) ? __uint_as_float(u ^ 0x80000000u) : __uint_as_float(~u);
}

// ---- WMMA fragment loaders (CDNA5 16x16x32 bf16 layouts, CDNA5 ISA) --
// A (16x32, MxK) from a row-major LDS tile: per lane the 16 values are two
// contiguous 8-element runs: K = k0+hi*8 .. +7 and K = k0+16+hi*8 .. +7.
// Both runs are 16B aligned -> two ds_load_b128.
static __device__ __forceinline__ v16bf load_a_frag(const bf16_t* sm, int ld, int row0, int k0) {
  const int lane = threadIdx.x & 31;
  const bf16_t* p = sm + (row0 + (lane & 15)) * ld + k0 + (lane >> 4) * 8;
  v8bf lo = *reinterpret_cast<const v8bf*>(p);
  v8bf hv = *reinterpret_cast<const v8bf*>(p + 16);
  v16bf a;
#pragma unroll
  for (int i = 0; i < 8; ++i) { a[i] = lo[i]; a[i + 8] = hv[i]; }
  return a;
}

// B fragments come from a frag-major ("swizzled") weight copy in LDS:
// idx(kt, nt, lane, v) = (((kt*ntiles + nt) * 32) + lane) * 16 + v
// so each lane's 16 values are contiguous and 32B aligned -> 2x ds_load_b128.
static __device__ __forceinline__ v16bf load_b_frag_swz(const bf16_t* sWs, int ntiles,
                                                        int kt, int nt) {
  const int lane = threadIdx.x & 31;
  return *reinterpret_cast<const v16bf*>(sWs + (((kt * ntiles + nt) << 5) + lane) * 16);
}

// Cooperative f32 -> bf16 staging of W[K,F] (row-major) into frag-major LDS.
// B layout per ISA: lane = hi*16 + n (n = column within 16-tile),
// value v<8 -> K = hi*8+v, v>=8 -> K = 16+hi*8+(v-8) (matches A fragment).
template <int K, int F, int NT>
static __device__ __forceinline__ void stage_weights_swz(const float* __restrict__ W,
                                                         bf16_t* sWs, int tid) {
  constexpr int NTILES = F >> 4;
  for (int i = tid; i < K * F; i += NT) {
    int k = i / F, n = i - k * F;         // F is a power of two, constexpr
    int kt = k >> 5, kr = k & 31;
    int idx = ((((kt * NTILES + (n >> 4)) << 5) + (((kr >> 3) & 1) << 4) + (n & 15)) << 4) + (kr & 7) + ((kr >> 4) << 3);
    sWs[idx] = (bf16_t)W[i];
  }
}

// ---------------------------------------------------------------------------
// Generic fills
// ---------------------------------------------------------------------------
__global__ void fill_f32_kernel(float* __restrict__ p, float v, int n) {
  int i = blockIdx.x * blockDim.x + threadIdx.x;
  if (i < n) p[i] = v;
}
__global__ void fill_u32_kernel(unsigned* __restrict__ p, unsigned v, int n) {
  int i = blockIdx.x * blockDim.x + threadIdx.x;
  if (i < n) p[i] = v;
}

// ---------------------------------------------------------------------------
// conv1 linear: H[N,64] = X[N,3] @ W1[3,64]   (K=3 -> scalar VALU)
// ---------------------------------------------------------------------------
__global__ void lin3_kernel(const float* __restrict__ X, const float* __restrict__ W,
                            float* __restrict__ H, int n) {
  int i = blockIdx.x * blockDim.x + threadIdx.x;
  if (i >= n * 64) return;
  int node = i >> 6, f = i & 63;
  float x0 = X[node * 3 + 0], x1 = X[node * 3 + 1], x2 = X[node * 3 + 2];
  H[i] = x0 * W[f] + x1 * W[64 + f] + x2 * W[128 + f];
}

// ---------------------------------------------------------------------------
// Node GEMM: H[n,F] = X[n,K] @ W[K,F]. bf16 WMMA, compile-time shapes.
// 128 threads (4 waves), 16 rows/wave, 64 rows/WG.
// ---------------------------------------------------------------------------
template <int K, int F>
__global__ __launch_bounds__(128)
void node_gemm_bf16_kernel(const float* __restrict__ X, const float* __restrict__ W,
                           float* __restrict__ H, int n) {
  constexpr int NTILES = F >> 4;
  constexpr int KTILES = K >> 5;
  __shared__ bf16_t sW[K * F];
  __shared__ bf16_t sX[64 * K];
  const int tid = threadIdx.x;
  const int row0 = blockIdx.x * 64;

  stage_weights_swz<K, F, 128>(W, sW, tid);
  for (int i = tid; i < 64 * K; i += 128) {
    int r = i / K, c = i - r * K;
    int gr = row0 + r;
    sX[i] = (gr < n) ? (bf16_t)X[gr * K + c] : (bf16_t)0.f;
  }
  __syncthreads();

  const int wv = tid >> 5;
  const int lane = tid & 31;
  const int hi = lane >> 4;
  const int nlo = lane & 15;
  const int gbase = row0 + wv * 16;
  const bool full = (row0 + 64 <= n);  // wave-uniform fast path

#pragma unroll
  for (int nt = 0; nt < NTILES; ++nt) {
    v8f acc = {};
#pragma unroll
    for (int kt = 0; kt < KTILES; ++kt) {
      v16bf a = load_a_frag(sX, K, wv * 16, kt << 5);
      v16bf b = load_b_frag_swz(sW, NTILES, kt, nt);
      acc = __builtin_amdgcn_wmma_f32_16x16x32_bf16(false, a, false, b, (short)0, acc,
                                                    false, false);
    }
    float* dstp = H + (size_t)(gbase + hi * 8) * F + (nt << 4) + nlo;
    if (full) {
#pragma unroll
      for (int r = 0; r < 8; ++r) dstp[(size_t)r * F] = acc[r];
    } else {
#pragma unroll
      for (int r = 0; r < 8; ++r)
        if (gbase + hi * 8 + r < n) dstp[(size_t)r * F] = acc[r];
    }
  }
}

// ---------------------------------------------------------------------------
// Fused EdgeConv: feats = cat[x_d, x_s - x_d] (128) -> relu(@We1+be1)
//                 -> relu(@We2+be2) -> atomicMax into Out[dst].
// 128 threads (4 waves), 64 edges/WG. LDS: swizzled weights 32KB (We1->We2),
// stage 16KB (feats->m1). Outputs >= 0, so uint atomicMax == float max.
// ---------------------------------------------------------------------------
__global__ __launch_bounds__(128)
void edgeconv_fused_kernel(const float* __restrict__ Hin,
                           const int* __restrict__ src, const int* __restrict__ dst,
                           const float* __restrict__ We1, const float* __restrict__ be1,
                           const float* __restrict__ We2, const float* __restrict__ be2,
                           unsigned* __restrict__ Out, int E) {
  __shared__ bf16_t sWgt[128 * 128];
  __shared__ bf16_t sStg[64 * 128];
  __shared__ float sB1[128];
  __shared__ float sB2[128];
  __shared__ int sDst[64];

  const int tid = threadIdx.x;
  const int e0 = blockIdx.x * 64;
  const bool fullE = (e0 + 64 <= E);

  sB1[tid] = be1[tid];
  sB2[tid] = be2[tid];
  if (tid < 64) { int e = e0 + tid; sDst[tid] = (e < E) ? dst[e] : 0; }
  stage_weights_swz<128, 128, 128>(We1, sWgt, tid);
  // Gather edge features (coalesced within a row: f == tid).
  for (int i = tid; i < 64 * 128; i += 128) {
    int row = i >> 7, f = i & 127;
    int e = e0 + row;
    float v = 0.f;
    if (e < E) {
      int d = dst[e];
      if (f < 64) {
        v = Hin[d * 64 + f];
      } else {
        int s = src[e];
        int f2 = f - 64;
        v = Hin[s * 64 + f2] - Hin[d * 64 + f2];
      }
    }
    sStg[i] = (bf16_t)v;
  }
  __syncthreads();

  const int wv = tid >> 5;
  const int lane = tid & 31;
  const int hi = lane >> 4;
  const int nlo = lane & 15;

  // Layer 1: keep all 8 N-tiles of the 16x128 result in VGPRs (64 regs).
  v8f accs[8];
#pragma unroll
  for (int nt = 0; nt < 8; ++nt) {
    v8f acc = {};
#pragma unroll
    for (int kt = 0; kt < 4; ++kt) {
      v16bf a = load_a_frag(sStg, 128, wv * 16, kt << 5);
      v16bf b = load_b_frag_swz(sWgt, 8, kt, nt);
      acc = __builtin_amdgcn_wmma_f32_16x16x32_bf16(false, a, false, b, (short)0, acc,
                                                    false, false);
    }
    accs[nt] = acc;
  }
  __syncthreads();

  // Swap weights, restage m1 = relu(layer1 + be1) as next A-matrix.
  stage_weights_swz<128, 128, 128>(We2, sWgt, tid);
#pragma unroll
  for (int nt = 0; nt < 8; ++nt) {
    int col = nt * 16 + nlo;
#pragma unroll
    for (int r = 0; r < 8; ++r) {
      int row = wv * 16 + hi * 8 + r;
      sStg[row * 128 + col] = (bf16_t)fmaxf(accs[nt][r] + sB1[col], 0.f);
    }
  }
  __syncthreads();

  // Layer 2 + scatter-max.
#pragma unroll
  for (int nt = 0; nt < 8; ++nt) {
    v8f acc = {};
#pragma unroll
    for (int kt = 0; kt < 4; ++kt) {
      v16bf a = load_a_frag(sStg, 128, wv * 16, kt << 5);
      v16bf b = load_b_frag_swz(sWgt, 8, kt, nt);
      acc = __builtin_amdgcn_wmma_f32_16x16x32_bf16(false, a, false, b, (short)0, acc,
                                                    false, false);
    }
    int col = nt * 16 + nlo;
    if (fullE) {
#pragma unroll
      for (int r = 0; r < 8; ++r) {
        int lrow = wv * 16 + hi * 8 + r;
        float v = fmaxf(acc[r] + sB2[col], 0.f);
        atomicMax(&Out[sDst[lrow] * 128 + col], __float_as_uint(v));
      }
    } else {
#pragma unroll
      for (int r = 0; r < 8; ++r) {
        int lrow = wv * 16 + hi * 8 + r;
        if (e0 + lrow < E) {
          float v = fmaxf(acc[r] + sB2[col], 0.f);
          atomicMax(&Out[sDst[lrow] * 128 + col], __float_as_uint(v));
        }
      }
    }
  }
}

// ---------------------------------------------------------------------------
// GAT attention (scalar / atomic path)
// ---------------------------------------------------------------------------
// Per-node alpha_src/alpha_dst dot products; one wave per node.
__global__ __launch_bounds__(256)
void alpha_kernel(const float* __restrict__ H, const float* __restrict__ aS,
                  const float* __restrict__ aD, float* __restrict__ As,
                  float* __restrict__ Ad, int n, int F) {
  int wave = (blockIdx.x * blockDim.x + threadIdx.x) >> 5;
  int lane = threadIdx.x & 31;
  if (wave >= n) return;
  float ps = 0.f, pd = 0.f;
  for (int f = lane; f < F; f += 32) {
    float h = H[wave * F + f];
    ps += h * aS[f];
    pd += h * aD[f];
  }
  for (int off = 16; off; off >>= 1) {
    ps += __shfl_xor(ps, off, 32);
    pd += __shfl_xor(pd, off, 32);
  }
  if (lane == 0) { As[wave] = ps; Ad[wave] = pd; }
}

// Pass 1: segment max of leaky-relu logits (self loops: idx >= E -> s=d=idx-E).
__global__ void logit_max_kernel(const int* __restrict__ src, const int* __restrict__ dst,
                                 const float* __restrict__ As, const float* __restrict__ Ad,
                                 unsigned* __restrict__ Mx, int E, int n) {
  int i = blockIdx.x * blockDim.x + threadIdx.x;
  if (i >= E + n) return;
  int s = (i < E) ? src[i] : (i - E);
  int d = (i < E) ? dst[i] : (i - E);
  atomicMax(&Mx[d], ord_enc(lrelu(As[s] + Ad[d])));
}

// Pass 2: e = exp(logit - m[d]); Sum[d] += e; Acc[d,:] += e * H[s,:].
__global__ __launch_bounds__(256)
void gat_accum_kernel(const int* __restrict__ src, const int* __restrict__ dst,
                      const float* __restrict__ As, const float* __restrict__ Ad,
                      const unsigned* __restrict__ Mx, const float* __restrict__ H,
                      float* __restrict__ Acc, float* __restrict__ Sum,
                      int E, int n, int F) {
  int wave = (blockIdx.x * blockDim.x + threadIdx.x) >> 5;
  int lane = threadIdx.x & 31;
  if (wave >= E + n) return;
  int s = (wave < E) ? src[wave] : (wave - E);
  int d = (wave < E) ? dst[wave] : (wave - E);
  float m = ord_dec(Mx[d]);
  float e = __expf(lrelu(As[s] + Ad[d]) - m);
  if (lane == 0) atomicAdd(&Sum[d], e);
  for (int f = lane; f < F; f += 32) atomicAdd(&Acc[d * F + f], e * H[s * F + f]);
}

// Finalize: Out = act(Acc / (Sum + 1e-16) + bias). act: 0=relu, 1=elu.
__global__ void gat_finalize_kernel(const float* __restrict__ Acc, const float* __restrict__ Sum,
                                    const float* __restrict__ bias, float* __restrict__ Out,
                                    int n, int Fshift, int act) {
  int i = blockIdx.x * blockDim.x + threadIdx.x;
  if (i >= (n << Fshift)) return;
  int node = i >> Fshift, f = i & ((1 << Fshift) - 1);
  float v = Acc[i] / (Sum[node] + 1e-16f) + bias[f];
  Out[i] = act ? (v > 0.f ? v : (__expf(v) - 1.f)) : fmaxf(v, 0.f);
}

// ---------------------------------------------------------------------------
// Pooling + head MLP
// ---------------------------------------------------------------------------
__global__ __launch_bounds__(256)
void pool_kernel(const float* __restrict__ H, const int* __restrict__ batch,
                 float* __restrict__ Sum, unsigned* __restrict__ Mx,
                 float* __restrict__ Cnt, int n) {
  int wave = (blockIdx.x * blockDim.x + threadIdx.x) >> 5;
  int lane = threadIdx.x & 31;
  if (wave >= n) return;
  int g = batch[wave];
  for (int f = lane; f < 128; f += 32) {
    float h = H[wave * 128 + f];  // h >= 0 (post-relu): uint max == float max
    atomicAdd(&Sum[g * 128 + f], h);
    atomicMax(&Mx[g * 128 + f], __float_as_uint(h));
  }
  if (lane == 0) atomicAdd(&Cnt[g], 1.f);
}

// z1 = relu([mean || max] @ Wm1 + bm1); thread per (g, f).
__global__ void pool_mlp_kernel(const float* __restrict__ Sum, const unsigned* __restrict__ Mx,
                                const float* __restrict__ Cnt, const float* __restrict__ Wm1,
                                const float* __restrict__ bm1, float* __restrict__ Z1) {
  int i = blockIdx.x * blockDim.x + threadIdx.x;
  if (i >= N_GRAPH * 128) return;
  int g = i >> 7, f = i & 127;
  float c = fmaxf(Cnt[g], 1.f);
  float a = bm1[f];
  for (int k = 0; k < 128; ++k) a += (Sum[g * 128 + k] / c) * Wm1[k * 128 + f];
  for (int k = 0; k < 128; ++k) a += __uint_as_float(Mx[g * 128 + k]) * Wm1[(128 + k) * 128 + f];
  Z1[i] = fmaxf(a, 0.f);
}

__global__ void mlp2_kernel(const float* __restrict__ Z1, const float* __restrict__ Wm2,
                            const float* __restrict__ bm2, float* __restrict__ Z2) {
  int i = blockIdx.x * blockDim.x + threadIdx.x;
  if (i >= N_GRAPH * 128) return;
  int g = i >> 7, f = i & 127;
  float a = bm2[f];
  for (int k = 0; k < 128; ++k) a += Z1[g * 128 + k] * Wm2[k * 128 + f];
  Z2[i] = fmaxf(a, 0.f);
}

__global__ void l2norm_kernel(const float* __restrict__ Z2, float* __restrict__ Out) {
  int g = blockIdx.x;
  int lane = threadIdx.x;
  float ss = 0.f;
  for (int f = lane; f < 128; f += 32) { float v = Z2[g * 128 + f]; ss += v * v; }
  for (int off = 16; off; off >>= 1) ss += __shfl_xor(ss, off, 32);
  float sc = 1.f / fmaxf(sqrtf(ss), 1e-12f);
  for (int f = lane; f < 128; f += 32) Out[g * 128 + f] = Z2[g * 128 + f] * sc;
}

// ---------------------------------------------------------------------------
// Host orchestration
// ---------------------------------------------------------------------------
static inline int cdiv(long a, long b) { return (int)((a + b - 1) / b); }

static void run_gat_attention(const float* H, int F, int Fshift, const int* src, const int* dst,
                              const float* aS, const float* aD, const float* bias,
                              float* Acc, float* Asb, float* Adb, unsigned* Mxb, float* Sumb,
                              float* Out, int act, hipStream_t stream) {
  const int N = N_NODES, E = N_EDGES;
  alpha_kernel<<<cdiv((long)N * 32, 256), 256, 0, stream>>>(H, aS, aD, Asb, Adb, N, F);
  fill_u32_kernel<<<cdiv(N, 256), 256, 0, stream>>>(Mxb, 0u, N);
  fill_f32_kernel<<<cdiv(N, 256), 256, 0, stream>>>(Sumb, 0.f, N);
  fill_f32_kernel<<<cdiv((long)N * F, 256), 256, 0, stream>>>(Acc, 0.f, N * F);
  logit_max_kernel<<<cdiv(E + N, 256), 256, 0, stream>>>(src, dst, Asb, Adb, Mxb, E, N);
  gat_accum_kernel<<<cdiv((long)(E + N) * 32, 256), 256, 0, stream>>>(src, dst, Asb, Adb, Mxb,
                                                                      H, Acc, Sumb, E, N, F);
  gat_finalize_kernel<<<cdiv((long)N * F, 256), 256, 0, stream>>>(Acc, Sumb, bias, Out, N,
                                                                  Fshift, act);
}

extern "C" void kernel_launch(void* const* d_in, const int* in_sizes, int n_in,
                              void* d_out, int out_size, void* d_ws, size_t ws_size,
                              hipStream_t stream) {
  (void)in_sizes; (void)n_in; (void)out_size; (void)ws_size;
  const int N = N_NODES, E = N_EDGES;

  const float* x   = (const float*)d_in[0];
  const int* ei    = (const int*)d_in[1];
  const int* batch = (const int*)d_in[2];
  const float* W1  = (const float*)d_in[3];
  const float* a1s = (const float*)d_in[4];
  const float* a1d = (const float*)d_in[5];
  const float* b1  = (const float*)d_in[6];
  const float* W2  = (const float*)d_in[7];
  const float* a2s = (const float*)d_in[8];
  const float* a2d = (const float*)d_in[9];
  const float* b2  = (const float*)d_in[10];
  const float* W3  = (const float*)d_in[11];
  const float* a3s = (const float*)d_in[12];
  const float* a3d = (const float*)d_in[13];
  const float* b3  = (const float*)d_in[14];
  const float* We1 = (const float*)d_in[15];
  const float* be1 = (const float*)d_in[16];
  const float* We2 = (const float*)d_in[17];
  const float* be2 = (const float*)d_in[18];
  const float* Wm1 = (const float*)d_in[19];
  const float* bm1 = (const float*)d_in[20];
  const float* Wm2 = (const float*)d_in[21];
  const float* bm2 = (const float*)d_in[22];
  const int* src = ei;
  const int* dst = ei + E;

  // Workspace carve-up (f32 elements).
  float* ws = (float*)d_ws;
  float* h64a  = ws; ws += (size_t)N * 64;
  float* h64b  = ws; ws += (size_t)N * 64;
  float* h128a = ws; ws += (size_t)N * 128;
  float* h128b = ws; ws += (size_t)N * 128;
  float* accb  = ws; ws += (size_t)N * 128;
  float* Asb   = ws; ws += N;
  float* Adb   = ws; ws += N;
  unsigned* Mxb = (unsigned*)ws; ws += N;
  float* Sumb  = ws; ws += N;
  float* pSum  = ws; ws += N_GRAPH * 128;
  unsigned* pMx = (unsigned*)ws; ws += N_GRAPH * 128;
  float* pCnt  = ws; ws += N_GRAPH;
  float* z1    = ws; ws += N_GRAPH * 128;
  float* z2    = ws; ws += N_GRAPH * 128;

  // ---- conv1 (GAT 3->64) + relu
  lin3_kernel<<<cdiv((long)N * 64, 256), 256, 0, stream>>>(x, W1, h64a, N);
  run_gat_attention(h64a, 64, 6, src, dst, a1s, a1d, b1, accb, Asb, Adb, Mxb, Sumb,
                    h64b, /*act=relu*/0, stream);

  // ---- conv2 (GAT 64->64) + elu
  node_gemm_bf16_kernel<64, 64><<<cdiv(N, 64), 128, 0, stream>>>(h64b, W2, h64a, N);
  run_gat_attention(h64a, 64, 6, src, dst, a2s, a2d, b2, accb, Asb, Adb, Mxb, Sumb,
                    h64b, /*act=elu*/1, stream);

  // ---- EdgeConv (64 -> 128), segment-max into h128a (init 0)
  fill_f32_kernel<<<cdiv((long)N * 128, 256), 256, 0, stream>>>(h128a, 0.f, N * 128);
  edgeconv_fused_kernel<<<cdiv(E, 64), 128, 0, stream>>>(h64b, src, dst, We1, be1, We2, be2,
                                                         (unsigned*)h128a, E);

  // ---- conv3 (GAT 128->128) + relu
  node_gemm_bf16_kernel<128, 128><<<cdiv(N, 64), 128, 0, stream>>>(h128a, W3, h128b, N);
  run_gat_attention(h128b, 128, 7, src, dst, a3s, a3d, b3, accb, Asb, Adb, Mxb, Sumb,
                    h128a, /*act=relu*/0, stream);

  // ---- global mean / max pool
  fill_f32_kernel<<<cdiv(N_GRAPH * 128, 256), 256, 0, stream>>>(pSum, 0.f, N_GRAPH * 128);
  fill_u32_kernel<<<cdiv(N_GRAPH * 128, 256), 256, 0, stream>>>(pMx, 0u, N_GRAPH * 128);
  fill_f32_kernel<<<1, 256, 0, stream>>>(pCnt, 0.f, N_GRAPH);
  pool_kernel<<<cdiv((long)N * 32, 256), 256, 0, stream>>>(h128a, batch, pSum, pMx, pCnt, N);

  // ---- head MLP + L2 normalize
  pool_mlp_kernel<<<cdiv(N_GRAPH * 128, 256), 256, 0, stream>>>(pSum, pMx, pCnt, Wm1, bm1, z1);
  mlp2_kernel<<<cdiv(N_GRAPH * 128, 256), 256, 0, stream>>>(z1, Wm2, bm2, z2);
  l2norm_kernel<<<N_GRAPH, 32, 0, stream>>>(z2, (float*)d_out);
}
